// PAM_Module_41652592836739
// MI455X (gfx1250) — matrix operand order repaired and checked by reading, not compiled, not run on hardware
//
#include <hip/hip_runtime.h>
#include <hip/hip_bf16.h>

#define BATCH 4
#define CH    512
#define CQ    64
#define NN    4096   // 64*64 spatial positions

typedef __attribute__((ext_vector_type(16))) __bf16 v16bf;
typedef __attribute__((ext_vector_type(8)))  float  v8f;

// native f32 -> bf16 truncation: lowers to v_cvt_pk_bf16_f32 (verified round 2)
__device__ __forceinline__ __bf16 f2bf(float f) { return (__bf16)f; }

// load 8 contiguous bf16 (16B) into fragment slots [e0, e0+8)
__device__ __forceinline__ void ld8(v16bf& f, int e0, const __bf16* p) {
  uint4 raw = *(const uint4*)p;
  const __bf16* r = (const __bf16*)&raw;
#pragma unroll
  for (int e = 0; e < 8; ++e) f[e0 + e] = r[e];
}

// load 16 contiguous bf16 (32B): B-matrix per-lane pattern (k = e + 16*half)
__device__ __forceinline__ v16bf ld16(const __bf16* p) {
  union { uint4 u[2]; v16bf v; } t;
  t.u[0] = *(const uint4*)p;
  t.u[1] = *(const uint4*)(p + 8);
  return t.v;
}

// CDNA5 async copy: 32B per lane, global -> LDS, tracked by ASYNCcnt.
// INST_OFFSET applies to both the LDS and global addresses (ISA 15.18.3 #98).
__device__ __forceinline__ void async_copy32(unsigned lds_off, unsigned voff,
                                             const void* sbase) {
  asm volatile("global_load_async_to_lds_b128 %0, %1, %2\n\t"
               "global_load_async_to_lds_b128 %0, %1, %2 offset:16"
               :: "v"(lds_off), "v"(voff),
                  "s"((unsigned long long)(size_t)sbase)
               : "memory");
}
__device__ __forceinline__ void wait_async0() {
  asm volatile("s_wait_asynccnt 0x0" ::: "memory");
}

// ---------------------------------------------------------------------------
// yT[b][n][o] = sum_c x[b][c][n] * W[o][c] + bias[o]   (O = CQ = 64, bf16 out)
// ---------------------------------------------------------------------------
__global__ void __launch_bounds__(256)
proj_qk_kernel(const float* __restrict__ x, const float* __restrict__ W,
               const float* __restrict__ bias, __bf16* __restrict__ yT)
{
  const int lane = threadIdx.x & 31;
  const int wave = threadIdx.x >> 5;
  const int half = lane >> 4;
  const int l15  = lane & 15;
  const int b    = blockIdx.z;
  const int n0   = blockIdx.x * 32 + (wave & 1) * 16;
  const int o0   = (wave >> 1) * 16;

  const float* xb = x + (size_t)b * CH * NN;

  v8f acc = {};
  for (int c0 = 0; c0 < CH; c0 += 32) {
    v16bf a;
#pragma unroll
    for (int e = 0; e < 16; ++e) {
      int k = 16 * half + e;
      a[e] = f2bf(xb[(size_t)(c0 + k) * NN + n0 + l15]);
    }
    const float* wp = W + (size_t)(o0 + l15) * CH + c0 + 16 * half;
    v16bf bfr;
#pragma unroll
    for (int e = 0; e < 16; ++e) bfr[e] = f2bf(wp[e]);
    acc = __builtin_amdgcn_wmma_f32_16x16x32_bf16(false, a, false, bfr, (short)0, acc, false, false);
  }
  const float bo = bias[o0 + l15];
  __bf16* yb = yT + (size_t)b * NN * CQ;
#pragma unroll
  for (int r = 0; r < 8; ++r) {
    int n = n0 + 8 * half + r;
    yb[(size_t)n * CQ + o0 + l15] = f2bf(acc[r] + bo);
  }
}

// ---------------------------------------------------------------------------
// y[b][o][n] = sum_c W[o][c] * x[b][c][n] + bias[o]   (O = CH = 512, bf16 out)
// ---------------------------------------------------------------------------
__global__ void __launch_bounds__(256)
proj_v_kernel(const float* __restrict__ x, const float* __restrict__ W,
              const float* __restrict__ bias, __bf16* __restrict__ y)
{
  const int lane = threadIdx.x & 31;
  const int wave = threadIdx.x >> 5;
  const int half = lane >> 4;
  const int l15  = lane & 15;
  const int b    = blockIdx.z;
  const int o0   = blockIdx.y * 32 + (wave & 1) * 16;
  const int n0   = blockIdx.x * 64 + (wave >> 1) * 16;

  const float* xb = x + (size_t)b * CH * NN;

  v8f acc = {};
  for (int c0 = 0; c0 < CH; c0 += 32) {
    const float* wp = W + (size_t)(o0 + l15) * CH + c0;
    v16bf a;
#pragma unroll
    for (int e = 0; e < 8; ++e)  a[e] = f2bf(wp[16 * half + e]);
#pragma unroll
    for (int e = 8; e < 16; ++e) a[e] = f2bf(wp[16 * half + e]);
    v16bf bfr;
#pragma unroll
    for (int e = 0; e < 16; ++e)
      bfr[e] = f2bf(xb[(size_t)(c0 + 16 * half + e) * NN + n0 + l15]);
    acc = __builtin_amdgcn_wmma_f32_16x16x32_bf16(false, a, false, bfr, (short)0, acc, false, false);
  }
  __bf16* yb = y + (size_t)b * CH * NN;
#pragma unroll
  for (int r = 0; r < 8; ++r) {
    int o = o0 + 8 * half + r;
    yb[(size_t)o * NN + n0 + l15] = f2bf(acc[r] + bias[o]);
  }
}

// ---------------------------------------------------------------------------
// Fused flash-attention + epilogue, 64 queries/block.
// K tiles double-buffered in LDS via global_load_async_to_lds_b128 (ASYNCcnt);
// the chunk-(j+1) DMA overlaps chunk-j softmax + P.V.
// ---------------------------------------------------------------------------
__global__ void __launch_bounds__(256)
attn_kernel(const __bf16* __restrict__ qT, const __bf16* __restrict__ kT,
            const __bf16* __restrict__ v,  const float* __restrict__ x,
            const float* __restrict__ gamma, float* __restrict__ out)
{
  __shared__ __align__(16) float  S_lds[64][64];
  __shared__ __align__(16) __bf16 P_lds[64][64];
  __shared__ __align__(16) __bf16 K_lds[2][64][64];   // double-buffered K tile
  __shared__ float m_s[64], l_s[64], scale_s[64];

  const int lane = threadIdx.x & 31;
  const int wave = threadIdx.x >> 5;
  const int half = lane >> 4;
  const int l15  = lane & 15;
  const int b    = blockIdx.y;
  const int i0   = blockIdx.x * 64;
  const int it   = wave & 3;        // i-subtile owned by this wave
  const int ch   = wave >> 2;       // c-half owned by this wave (0 or 1)
  const int cbase = ch * 256;
  const unsigned tid = threadIdx.x;

  const __bf16* qb = qT + (size_t)b * NN * CQ;
  const __bf16* kb = kT + (size_t)b * NN * CQ;
  const __bf16* vb = v  + (size_t)b * CH * NN;

  // Q A-fragments for this wave's i-subtile (row-contiguous in qT)
  v16bf aq[2];
#pragma unroll
  for (int c = 0; c < 2; ++c) {
    const __bf16* qp = qb + (size_t)(i0 + it * 16 + l15) * CQ + c * 32;
    ld8(aq[c], 0, qp + 16 * half);
    ld8(aq[c], 8, qp + 16 * half + 8);
  }

  v8f acc[16] = {};
  if (tid < 64) { m_s[tid] = -1e30f; l_s[tid] = 0.0f; }

  // prologue: stage K tile for j0 = 0 into buffer 0 (8KB = 256 thr x 32B)
  async_copy32((unsigned)(size_t)&K_lds[0][0][0] + tid * 32u, tid * 32u, kb);

  for (int j0 = 0; j0 < NN; j0 += 64) {
    const int buf = (j0 >> 6) & 1;

    if (j0 + 64 < NN)
      __builtin_prefetch(vb + (size_t)(cbase + lane * 8) * NN + (j0 + 64), 0, 1);

    wait_async0();
    __syncthreads();                    // (1) K tile[buf] visible to all waves

    // ---- S tiles: wave does (it, jtl) for jtl = (w>>2)*2 + {0,1} ----
#pragma unroll
    for (int t = 0; t < 2; ++t) {
      const int jtl = (wave >> 2) * 2 + t;
      v8f s = {};
#pragma unroll
      for (int c = 0; c < 2; ++c) {
        v16bf bk = ld16(&K_lds[buf][jtl * 16 + l15][c * 32 + 16 * half]);
        s = __builtin_amdgcn_wmma_f32_16x16x32_bf16(false, aq[c], false, bk, (short)0, s, false, false);
      }
#pragma unroll
      for (int r = 0; r < 8; ++r)
        S_lds[16 * it + 8 * half + r][jtl * 16 + l15] = s[r];
    }
    __syncthreads();                    // (2) S complete; K[buf] reads done

    // kick off DMA of the next K tile into the other buffer (overlaps below)
    if (j0 + 64 < NN)
      async_copy32((unsigned)(size_t)&K_lds[buf ^ 1][0][0] + tid * 32u,
                   tid * 32u, kb + (size_t)(j0 + 64) * CQ);

    // ---- online softmax: wave w handles rows [w*8, w*8+8) ----
    {
      const int row = wave * 8 + (lane >> 2);
      const int q   = lane & 3;
      float mold = m_s[row], lold = l_s[row];
      float mx = -1e30f;
#pragma unroll
      for (int t = 0; t < 16; ++t) mx = fmaxf(mx, S_lds[row][q + 4 * t]);
      mx = fmaxf(mx, __shfl_xor(mx, 1, 32));
      mx = fmaxf(mx, __shfl_xor(mx, 2, 32));
      float mnew = fmaxf(mold, mx);
      float sum = 0.0f;
#pragma unroll
      for (int t = 0; t < 16; ++t) {
        float p = __expf(S_lds[row][q + 4 * t] - mnew);
        P_lds[row][q + 4 * t] = f2bf(p);
        sum += p;
      }
      sum += __shfl_xor(sum, 1, 32);
      sum += __shfl_xor(sum, 2, 32);
      if (q == 0) {
        float sc = __expf(mold - mnew);
        scale_s[row] = sc;
        m_s[row]     = mnew;
        l_s[row]     = lold * sc + sum;
      }
    }
    __syncthreads();                    // (3) P/stats ready

    // ---- rescale running accumulators ----
    float sc[8];
#pragma unroll
    for (int r = 0; r < 8; ++r) sc[r] = scale_s[16 * it + 8 * half + r];
#pragma unroll
    for (int t = 0; t < 16; ++t)
#pragma unroll
      for (int r = 0; r < 8; ++r) acc[t][r] *= sc[r];

    // ---- acc += P[16i x 64j] * V^T[64j x 256c], K chunks of 32 ----
#pragma unroll
    for (int kc = 0; kc < 2; ++kc) {
      v16bf ap;
      const __bf16* pp = &P_lds[16 * it + l15][kc * 32];
      ld8(ap, 0, pp + 16 * half);
      ld8(ap, 8, pp + 16 * half + 8);
#pragma unroll
      for (int ct = 0; ct < 16; ++ct) {
        const __bf16* vp = vb + (size_t)(cbase + ct * 16 + l15) * NN
                              + j0 + kc * 32 + 16 * half;
        v16bf bv = ld16(vp);
        acc[ct] = __builtin_amdgcn_wmma_f32_16x16x32_bf16(false, ap, false, bv, (short)0, acc[ct], false, false);
      }
    }
    // no trailing barrier needed: next-iteration writers of S_lds/P_lds/stats
    // are all gated behind barrier (1)/(2) of the next iteration.
  }

  // ---- epilogue: out = gamma * (acc / l) + x ----
  const float g = gamma[0];
  float invl[8];
#pragma unroll
  for (int r = 0; r < 8; ++r) invl[r] = 1.0f / l_s[16 * it + 8 * half + r];
  const float* xb = x   + (size_t)b * CH * NN;
  float*       ob = out + (size_t)b * CH * NN;
#pragma unroll
  for (int ct = 0; ct < 16; ++ct) {
#pragma unroll
    for (int r = 0; r < 8; ++r) {
      int i = i0 + it * 16 + 8 * half + r;
      int c = cbase + ct * 16 + l15;
      size_t idx = (size_t)c * NN + i;
      ob[idx] = g * (acc[ct][r] * invl[r]) + xb[idx];
    }
  }
}

extern "C" void kernel_launch(void* const* d_in, const int* in_sizes, int n_in,
                              void* d_out, int out_size, void* d_ws, size_t ws_size,
                              hipStream_t stream) {
  (void)in_sizes; (void)n_in; (void)out_size; (void)ws_size;
  const float* x     = (const float*)d_in[0];
  const float* Wq    = (const float*)d_in[1];
  const float* bq    = (const float*)d_in[2];
  const float* Wk    = (const float*)d_in[3];
  const float* bk    = (const float*)d_in[4];
  const float* Wv    = (const float*)d_in[5];
  const float* bv    = (const float*)d_in[6];
  const float* gamma = (const float*)d_in[7];

  // workspace: qT [B,N,64] + kT [B,N,64] + v [B,512,N], all bf16 (~20 MB)
  __bf16* qT = (__bf16*)d_ws;
  __bf16* kT = qT + (size_t)BATCH * NN * CQ;
  __bf16* vv = kT + (size_t)BATCH * NN * CQ;

  proj_qk_kernel<<<dim3(NN / 32, 1, BATCH), 256, 0, stream>>>(x, Wq, bq, qT);
  proj_qk_kernel<<<dim3(NN / 32, 1, BATCH), 256, 0, stream>>>(x, Wk, bk, kT);
  proj_v_kernel <<<dim3(NN / 64, CH / 32, BATCH), 256, 0, stream>>>(x, Wv, bv, vv);
  attn_kernel   <<<dim3(NN / 64, BATCH), 256, 0, stream>>>(qT, kT, vv, x, gamma, (float*)d_out);
}
